// RegionalGNN_50603304682248
// MI455X (gfx1250) — compile-verified, matrix operand order checked by reading, not run on hardware
//
#include <hip/hip_runtime.h>
#include <cstdint>
#include <cstddef>

// ---------- types ----------
typedef __attribute__((ext_vector_type(16))) __bf16 bf16x16;
typedef __attribute__((ext_vector_type(8)))  float  f32x8;
typedef __attribute__((ext_vector_type(4)))  unsigned int u32x4;

#define NL 256        // latent width
#define KMAX 768      // max padded K (3*latent)

union FragU { u32x4 q[2]; bf16x16 v; };

// Load a 16-value bf16 A/B fragment slice for this lane.
// Per ISA layout, lane needs K = {kb..kb+7} and {kb+16..kb+23} (kb already
// includes the lane>=16 half-shift of +8). p points at element kb of the row.
__device__ __forceinline__ bf16x16 frag_ld(const __bf16* p) {
  FragU f;
  const u32x4* q = reinterpret_cast<const u32x4*>(p);
  f.q[0] = q[0];   // elements kb .. kb+7
  f.q[1] = q[2];   // elements kb+16 .. kb+23
  return f.v;
}

// ---------- fused 2-layer MLP (swish hidden) + LayerNorm + residual ----------
// A: bf16 [Mpad16][Kpad] row-major (zero padded). W1T: bf16 [256][Kpad],
// W2T: bf16 [256][256] (transposed: row = output col, contiguous K).
// out = res + LN(W2^T swish(W1^T a + b1) + b2) * gamma + beta
__global__ __launch_bounds__(128)
void mlp2_ln_kernel(const __bf16* __restrict__ A, int Kpad,
                    const __bf16* __restrict__ W1T, const float* __restrict__ b1,
                    const __bf16* __restrict__ W2T, const float* __restrict__ b2,
                    const float* __restrict__ gamma, const float* __restrict__ beta,
                    const float* __restrict__ res, float* __restrict__ out, int M)
{
  constexpr int HSTR = NL + 8;                 // bf16 hidden stride (bank-conflict pad)
  __shared__ __bf16 Ash[16 * KMAX];            // staged A tile (async DMA target)
  __shared__ __bf16 Hsh[16 * HSTR];
  __shared__ float  Ysh[16 * NL];

  const int lane  = threadIdx.x & 31;
  const int wave  = threadIdx.x >> 5;          // 4 waves; wave owns cols [64w,64w+64)
  const int r     = lane & 15;                 // row (A) / col (B) within tile
  const int khalf = (lane >> 4) << 3;          // 0 or 8
  const long long row0 = (long long)blockIdx.x * 16;

  // ---- async-stage the contiguous 16 x Kpad A tile into LDS ----
  {
    const char* gbase = (const char*)(A + row0 * (long long)Kpad);
    const unsigned ldsBase = (unsigned)(uintptr_t)(&Ash[0]);   // addr[31:0] = LDS offset
    const int tileBytes = 16 * Kpad * 2;
    for (int off = threadIdx.x * 16; off < tileBytes; off += 128 * 16) {
      unsigned ldsA = ldsBase + (unsigned)off;
      const char* ga = gbase + off;
      asm volatile("global_load_async_to_lds_b128 %0, %1, off"
                   :: "v"(ldsA), "v"(ga) : "memory");
    }
    asm volatile("s_wait_asynccnt 0x0" ::: "memory");
  }
  __syncthreads();

  const f32x8 z = {0.f,0.f,0.f,0.f,0.f,0.f,0.f,0.f};
  f32x8 acc[4] = {z, z, z, z};

  // ---- layer 1: [16 x Kpad] x [Kpad x 256] ----
  const int nk1 = Kpad >> 5;
  for (int kt = 0; kt < nk1; ++kt) {
    const int kb = (kt << 5) + khalf;
    bf16x16 a = frag_ld(&Ash[r * Kpad + kb]);          // ds_load_b128 x2
    bf16x16 bf[4];
#pragma unroll
    for (int t = 0; t < 4; ++t) {                      // issue all B loads first
      const int n = wave * 64 + t * 16 + r;
      bf[t] = frag_ld(W1T + (long long)n * Kpad + kb);
    }
#pragma unroll
    for (int t = 0; t < 4; ++t)
      acc[t] = __builtin_amdgcn_wmma_f32_16x16x32_bf16(
                 false, a, false, bf[t], (short)0, acc[t], false, false);
  }

  // bias + swish -> bf16 hidden in LDS (C layout: vgpr j -> row j+khalf, col = l&15)
#pragma unroll
  for (int t = 0; t < 4; ++t) {
    const int col = wave * 64 + t * 16 + r;
    const float bb = b1[col];
#pragma unroll
    for (int j = 0; j < 8; ++j) {
      float x = acc[t][j] + bb;
      float sw = x / (1.f + __expf(-x));
      Hsh[(j + khalf) * HSTR + col] = (__bf16)sw;
    }
  }
  __syncthreads();

  // ---- layer 2: [16 x 256] x [256 x 256] ----
  f32x8 acc2[4] = {z, z, z, z};
#pragma unroll
  for (int kt = 0; kt < 8; ++kt) {
    const int kb = (kt << 5) + khalf;
    bf16x16 a = frag_ld(&Hsh[r * HSTR + kb]);
    bf16x16 bf[4];
#pragma unroll
    for (int t = 0; t < 4; ++t) {
      const int n = wave * 64 + t * 16 + r;
      bf[t] = frag_ld(W2T + n * NL + kb);
    }
#pragma unroll
    for (int t = 0; t < 4; ++t)
      acc2[t] = __builtin_amdgcn_wmma_f32_16x16x32_bf16(
                  false, a, false, bf[t], (short)0, acc2[t], false, false);
  }

#pragma unroll
  for (int t = 0; t < 4; ++t) {
    const int col = wave * 64 + t * 16 + r;
    const float bb = b2[col];
#pragma unroll
    for (int j = 0; j < 8; ++j)
      Ysh[(j + khalf) * NL + col] = acc2[t][j] + bb;
  }
  __syncthreads();

  // ---- LayerNorm: wave handles rows [4w, 4w+4); lane covers 8 cols ----
  for (int rr = 0; rr < 4; ++rr) {
    const int row = wave * 4 + rr;
    const float4* yrow = reinterpret_cast<const float4*>(&Ysh[row * NL]);
    float4 v0 = yrow[lane * 2 + 0];
    float4 v1 = yrow[lane * 2 + 1];
    float s  = v0.x + v0.y + v0.z + v0.w + v1.x + v1.y + v1.z + v1.w;
    float sq = v0.x*v0.x + v0.y*v0.y + v0.z*v0.z + v0.w*v0.w
             + v1.x*v1.x + v1.y*v1.y + v1.z*v1.z + v1.w*v1.w;
#pragma unroll
    for (int m = 16; m >= 1; m >>= 1) {
      s  += __shfl_xor(s,  m, 32);
      sq += __shfl_xor(sq, m, 32);
    }
    const float mean = s * (1.f / 256.f);
    const float var  = sq * (1.f / 256.f) - mean * mean;   // biased, matches jnp.var
    const float rstd = rsqrtf(var + 1e-5f);
    const long long grow = row0 + row;
    if (grow < M) {
      float* op = out + grow * (long long)NL;
      const float* rp = res ? res + grow * (long long)NL : nullptr;
      const float vals[8] = {v0.x, v0.y, v0.z, v0.w, v1.x, v1.y, v1.z, v1.w};
      const int c0 = lane * 8;
#pragma unroll
      for (int i = 0; i < 8; ++i) {
        const int c = c0 + i;
        float y = (vals[i] - mean) * rstd * gamma[c] + beta[c];
        if (rp) y += rp[c];
        op[c] = y;
      }
    }
  }
}

// ---------- packing / gather / aggregation kernels ----------

// f32 [Mreal][Kreal] (starting at srcRowOff) -> bf16 [ceil16(M)][Kpad], zero padded
__global__ void pack_rows_bf16(const float* __restrict__ src, __bf16* __restrict__ dst,
                               int Mreal, int Kreal, int Kpad, int srcRowOff,
                               long long total)
{
  long long idx = (long long)blockIdx.x * blockDim.x + threadIdx.x;
  if (idx >= total) return;
  int k = (int)(idx % Kpad);
  long long row = idx / Kpad;
  float v = 0.f;
  if (row < Mreal && k < Kreal)
    v = src[((long long)srcRowOff + row) * Kreal + k];
  dst[idx] = (__bf16)v;
}

// W [K][256] f32 -> WT [256][Kpad] bf16 (transpose + pad)
__global__ void pack_w_t(const float* __restrict__ w, __bf16* __restrict__ dst,
                         int K, int Kpad)
{
  int idx = blockIdx.x * blockDim.x + threadIdx.x;
  if (idx >= 256 * Kpad) return;
  int k = idx % Kpad, n = idx / Kpad;
  float v = (k < K) ? w[(long long)k * 256 + n] : 0.f;
  dst[idx] = (__bf16)v;
}

// concat([e, nodeS[send], nodeR[recv]]) -> bf16 [E][768]
__global__ void concat_edge(const float* __restrict__ e,
                            const float* __restrict__ nodeS, const int* __restrict__ send,
                            const float* __restrict__ nodeR, const int* __restrict__ recv,
                            __bf16* __restrict__ dst, int E, long long total)
{
  long long idx = (long long)blockIdx.x * blockDim.x + threadIdx.x;
  if (idx >= total) return;
  int c = (int)(idx % 768);
  long long ed = idx / 768;
  float v = 0.f;
  if (ed < E) {
    if (c < 256)      v = e[ed * 256 + c];
    else if (c < 512) v = nodeS[(long long)send[ed] * 256 + (c - 256)];
    else              v = nodeR[(long long)recv[ed] * 256 + (c - 512)];
  }
  dst[idx] = (__bf16)v;
}

// concat([x, a, b]) -> bf16 [ceil16(M)][768], zero padded rows
__global__ void concat_node(const float* __restrict__ x, const float* __restrict__ a,
                            const float* __restrict__ b,
                            __bf16* __restrict__ dst, int M, long long total)
{
  long long idx = (long long)blockIdx.x * blockDim.x + threadIdx.x;
  if (idx >= total) return;
  int c = (int)(idx % 768);
  long long row = idx / 768;
  float v = 0.f;
  if (row < M) {
    if (c < 256)      v = x[row * 256 + c];
    else if (c < 512) v = a[row * 256 + (c - 256)];
    else              v = b[row * 256 + (c - 512)];
  }
  dst[idx] = (__bf16)v;
}

__global__ void zero_f32(float* __restrict__ p, long long n)
{
  long long idx = (long long)blockIdx.x * blockDim.x + threadIdx.x;
  if (idx < n) p[idx] = 0.f;
}

// segment_sum: agg[recv[e]] += e_state[e]
__global__ void scatter_add(const float* __restrict__ e, const int* __restrict__ recv,
                            float* __restrict__ agg, long long total)
{
  long long idx = (long long)blockIdx.x * blockDim.x + threadIdx.x;
  if (idx >= total) return;
  int c = (int)(idx & 255);
  long long ed = idx >> 8;
  atomicAdd(&agg[(long long)recv[ed] * 256 + c], e[idx]);
}

// decoder: out[m] = sum_t swish(dot(down[m], W1[:,t]) + b1[t]) * w2[t] + b2
__global__ __launch_bounds__(256)
void decoder_kernel(const float* __restrict__ down, const float* __restrict__ w1,
                    const float* __restrict__ b1, const float* __restrict__ w2,
                    const float* __restrict__ b2, float* __restrict__ out)
{
  __shared__ float xs[256];
  __shared__ float red[256];
  const long long m = blockIdx.x;
  const int t = threadIdx.x;
  xs[t] = down[m * 256 + t];
  __syncthreads();
  float acc = b1[t];
  for (int k = 0; k < 256; ++k) acc = fmaf(xs[k], w1[k * 256 + t], acc);
  float h = acc / (1.f + __expf(-acc));
  red[t] = h * w2[t];
  __syncthreads();
  for (int off = 128; off > 0; off >>= 1) {
    if (t < off) red[t] += red[t + off];
    __syncthreads();
  }
  if (t == 0) out[m] = red[0] + b2[0];
}

// ---------- host ----------
struct BlockW { const float *ln_off, *ln_scl, *b1, *w1, *b2, *w2; };

static inline unsigned cdiv_ll(long long a, long long b) { return (unsigned)((a + b - 1) / b); }

extern "C" void kernel_launch(void* const* d_in, const int* in_sizes, int n_in,
                              void* d_out, int out_size, void* d_ws, size_t ws_size,
                              hipStream_t stream)
{
  (void)in_sizes; (void)n_in; (void)out_size; (void)ws_size;
  constexpr int kNUp = 25000, kNDown = 12500, kE = 50000;
  constexpr int kNUpP = 25008, kNDownP = 12512;            // ceil16
  constexpr int kFNode = 336, kFNodeP = 352, kFEdge = 4, kFEdgeP = 32;

  const float* node_feat = (const float*)d_in[0];
  const float* ei_feat   = (const float*)d_in[1];
  const float* eu_feat   = (const float*)d_in[2];
  const int* intra_send  = (const int*)d_in[3];
  const int* intra_recv  = (const int*)d_in[4];
  const int* u2d_send    = (const int*)d_in[5];
  const int* u2d_recv    = (const int*)d_in[6];

  // params flattened in jax tree order (dict keys sorted):
  // decoder(4), embed_down(6), embed_e_intra(6), embed_e_u2d(6), embed_up(6),
  // steps[3] x {edge_intra, edge_u2d, node_down, node_up} x 6
  const float* dec_b1 = (const float*)d_in[7];
  const float* dec_w1 = (const float*)d_in[8];
  const float* dec_b2 = (const float*)d_in[9];
  const float* dec_w2 = (const float*)d_in[10];
  auto blk = [&](int base) {
    BlockW b;
    b.ln_off = (const float*)d_in[base + 0];
    b.ln_scl = (const float*)d_in[base + 1];
    b.b1     = (const float*)d_in[base + 2];
    b.w1     = (const float*)d_in[base + 3];
    b.b2     = (const float*)d_in[base + 4];
    b.w2     = (const float*)d_in[base + 5];
    return b;
  };
  BlockW blocks[16]; int kp[16], kr[16];
  blocks[0] = blk(29); kp[0] = kFNodeP; kr[0] = kFNode;   // embed_up
  blocks[1] = blk(11); kp[1] = kFNodeP; kr[1] = kFNode;   // embed_down
  blocks[2] = blk(17); kp[2] = kFEdgeP; kr[2] = kFEdge;   // embed_e_intra
  blocks[3] = blk(23); kp[3] = kFEdgeP; kr[3] = kFEdge;   // embed_e_u2d
  for (int s = 0; s < 3; ++s)
    for (int j = 0; j < 4; ++j) {                         // ei, eu, nd, nu
      int i = 4 + 4 * s + j;
      blocks[i] = blk(35 + s * 24 + j * 6);
      kp[i] = (j == 3) ? 256 : 768;
      kr[i] = (j == 3) ? 256 : 768;
    }

  // workspace carving
  char* ws = (char*)d_ws;
  size_t cur = 0;
  auto carve = [&](size_t bytes) -> char* {
    char* p = ws + cur;
    cur += (bytes + 255) & ~(size_t)255;
    return p;
  };
  __bf16* Aup   = (__bf16*)carve((size_t)kNUpP * kFNodeP * 2);
  __bf16* Adown = (__bf16*)carve((size_t)kNDownP * kFNodeP * 2);
  __bf16* Aei   = (__bf16*)carve((size_t)kE * kFEdgeP * 2);
  __bf16* Aeu   = (__bf16*)carve((size_t)kE * kFEdgeP * 2);
  float* up   = (float*)carve((size_t)kNUp   * NL * 4);
  float* down = (float*)carve((size_t)kNDown * NL * 4);
  float* eInt = (float*)carve((size_t)kE * NL * 4);
  float* eU2d = (float*)carve((size_t)kE * NL * 4);
  float* aggI = (float*)carve((size_t)kNDown * NL * 4);
  float* aggU = (float*)carve((size_t)kNDown * NL * 4);
  __bf16* Cat = (__bf16*)carve((size_t)kE * 768 * 2);
  __bf16 *w1t[16], *w2t[16];
  for (int i = 0; i < 16; ++i) {
    w1t[i] = (__bf16*)carve((size_t)256 * kp[i] * 2);
    w2t[i] = (__bf16*)carve((size_t)256 * 256 * 2);
  }

  // pack all weights (bf16, transposed, K padded)
  for (int i = 0; i < 16; ++i) {
    pack_w_t<<<cdiv_ll(256LL * kp[i], 256), 256, 0, stream>>>(blocks[i].w1, w1t[i], kr[i], kp[i]);
    pack_w_t<<<cdiv_ll(256LL * 256, 256),   256, 0, stream>>>(blocks[i].w2, w2t[i], 256, 256);
  }
  // pack encoder inputs
  {
    long long t;
    t = (long long)kNUpP * kFNodeP;
    pack_rows_bf16<<<cdiv_ll(t, 256), 256, 0, stream>>>(node_feat, Aup, kNUp, kFNode, kFNodeP, 0, t);
    t = (long long)kNDownP * kFNodeP;
    pack_rows_bf16<<<cdiv_ll(t, 256), 256, 0, stream>>>(node_feat, Adown, kNDown, kFNode, kFNodeP, kNUp, t);
    t = (long long)kE * kFEdgeP;
    pack_rows_bf16<<<cdiv_ll(t, 256), 256, 0, stream>>>(ei_feat, Aei, kE, kFEdge, kFEdgeP, 0, t);
    pack_rows_bf16<<<cdiv_ll(t, 256), 256, 0, stream>>>(eu_feat, Aeu, kE, kFEdge, kFEdgeP, 0, t);
  }

  auto runMLP = [&](int i, const __bf16* Abuf, int Kpad, int Mpad, int M,
                    const float* res, float* out) {
    mlp2_ln_kernel<<<Mpad / 16, 128, 0, stream>>>(Abuf, Kpad, w1t[i], blocks[i].b1,
                                                  w2t[i], blocks[i].b2,
                                                  blocks[i].ln_scl, blocks[i].ln_off,
                                                  res, out, M);
  };

  // ---- encoder ----
  runMLP(0, Aup,   kFNodeP, kNUpP,   kNUp,   nullptr, up);
  runMLP(1, Adown, kFNodeP, kNDownP, kNDown, nullptr, down);
  runMLP(2, Aei,   kFEdgeP, kE,      kE,     nullptr, eInt);
  runMLP(3, Aeu,   kFEdgeP, kE,      kE,     nullptr, eU2d);

  // ---- processor steps ----
  for (int s = 0; s < 3; ++s) {
    const int bi = 4 + 4 * s;
    long long tE = (long long)kE * 768;
    // edge_intra
    concat_edge<<<cdiv_ll(tE, 256), 256, 0, stream>>>(eInt, down, intra_send, down, intra_recv, Cat, kE, tE);
    runMLP(bi + 0, Cat, 768, kE, kE, eInt, eInt);
    // edge_u2d
    concat_edge<<<cdiv_ll(tE, 256), 256, 0, stream>>>(eU2d, up, u2d_send, down, u2d_recv, Cat, kE, tE);
    runMLP(bi + 1, Cat, 768, kE, kE, eU2d, eU2d);
    // aggregation
    long long nD = (long long)kNDown * NL;
    zero_f32<<<cdiv_ll(nD, 256), 256, 0, stream>>>(aggI, nD);
    zero_f32<<<cdiv_ll(nD, 256), 256, 0, stream>>>(aggU, nD);
    long long tS = (long long)kE * NL;
    scatter_add<<<cdiv_ll(tS, 256), 256, 0, stream>>>(eInt, intra_recv, aggI, tS);
    scatter_add<<<cdiv_ll(tS, 256), 256, 0, stream>>>(eU2d, u2d_recv, aggU, tS);
    // node_down
    long long tN = (long long)kNDownP * 768;
    concat_node<<<cdiv_ll(tN, 256), 256, 0, stream>>>(down, aggI, aggU, Cat, kNDown, tN);
    runMLP(bi + 2, Cat, 768, kNDownP, kNDown, down, down);
    // node_up (pack current up to bf16 first; update is in-place afterwards)
    long long tU = (long long)kNUpP * NL;
    pack_rows_bf16<<<cdiv_ll(tU, 256), 256, 0, stream>>>(up, Cat, kNUp, NL, NL, 0, tU);
    runMLP(bi + 3, Cat, 256, kNUpP, kNUp, up, up);
  }

  // ---- decoder ----
  decoder_kernel<<<kNDown, 256, 0, stream>>>(down, dec_w1, dec_b1, dec_w2, dec_b2, (float*)d_out);
}
